// NTXentLoss_74019466379880
// MI455X (gfx1250) — matrix operand order repaired and checked by reading, not compiled, not run on hardware
//
#include <hip/hip_runtime.h>
#include <hip/hip_bf16.h>

// ---------------- problem constants ----------------
#define BATCH    4096
#define NROWS    8192          // 2*BATCH
#define DIM      256
#define KCHUNKS  8             // 256 / 32 (K per wmma)
#define NCHUNK   8             // column-range split per row block
#define COLTILES_PER_CHUNK 64  // (8192/16)/NCHUNK
#define INV_TEMP 2.0f          // 1/0.5
#define NEG_INF  (-1000000000.0f)

typedef __attribute__((ext_vector_type(16))) _Float16 v16h;
typedef __attribute__((ext_vector_type(8)))  _Float16 v8h;
typedef __attribute__((ext_vector_type(8)))  float    v8f;

// ---------------- kernel 1: zero per-row accumulators ----------------
__global__ __launch_bounds__(256)
void ntx_zero_kernel(float* __restrict__ rowsum) {
    int i = blockIdx.x * blockDim.x + threadIdx.x;
    if (i < NROWS) rowsum[i] = 0.0f;
}

// ---------------- kernel 2: row-normalize, fp32 -> f16 ----------------
// one wave per row; lane handles 8 contiguous elements
__global__ __launch_bounds__(256)
void ntx_normalize_kernel(const float* __restrict__ z1,
                          const float* __restrict__ z2,
                          _Float16* __restrict__ z16) {
    const int wave = (blockIdx.x * blockDim.x + threadIdx.x) >> 5;
    const int lane = threadIdx.x & 31;
    if (wave >= NROWS) return;
    const float* src = (wave < BATCH) ? (z1 + (size_t)wave * DIM)
                                      : (z2 + (size_t)(wave - BATCH) * DIM);
    const float4 p0 = *(const float4*)(src + lane * 8);
    const float4 p1 = *(const float4*)(src + lane * 8 + 4);
    float ss = p0.x*p0.x + p0.y*p0.y + p0.z*p0.z + p0.w*p0.w
             + p1.x*p1.x + p1.y*p1.y + p1.z*p1.z + p1.w*p1.w;
    #pragma unroll
    for (int m = 1; m < 32; m <<= 1) ss += __shfl_xor(ss, m, 32);
    float nrm = sqrtf(ss);
    nrm = fmaxf(nrm, 1e-12f);
    const float inv = 1.0f / nrm;
    v8h out;
    out[0] = (_Float16)(p0.x * inv); out[1] = (_Float16)(p0.y * inv);
    out[2] = (_Float16)(p0.z * inv); out[3] = (_Float16)(p0.w * inv);
    out[4] = (_Float16)(p1.x * inv); out[5] = (_Float16)(p1.y * inv);
    out[6] = (_Float16)(p1.z * inv); out[7] = (_Float16)(p1.w * inv);
    *(v8h*)(z16 + (size_t)wave * DIM + lane * 8) = out;
}

// ---------------- kernel 3: fused WMMA Gram sweep + exp-sum ----------------
// grid = (512 row blocks, NCHUNK column chunks), 1 wave per block.
// A panel: 16 rows x K=256 held in 8 v16h frags for the whole sweep.
// 16-bit A-frag VGPR swizzle (ISA 7.12.2): lane (h=lane>>4, r=lane&15) holds
// row m0+r, element e maps to K = kbase + e + 8*h + (e>=8 ? 8 : 0).
// B tile (Gram: B = Z^T): lane (h,r) holds column n0+r, same K mapping as A.
// B tiles are double-buffered: full 8 KB tile (16 x b128) prefetched while the
// 8 WMMAs of the current tile execute.
__global__ __launch_bounds__(32)
void ntx_gemm_rowsum_kernel(const _Float16* __restrict__ z16,
                            float* __restrict__ rowsum) {
    const int lane = threadIdx.x;
    const int r = lane & 15;
    const int h = lane >> 4;
    const int m0 = blockIdx.x * 16;

    // load A fragments (held across the column sweep)
    v16h afrag[KCHUNKS];
    const _Float16* arow = z16 + (size_t)(m0 + r) * DIM;
    #pragma unroll
    for (int kc = 0; kc < KCHUNKS; ++kc) {
        const int kb = kc * 32;
        const v8h lo = *(const v8h*)(arow + kb + 8 * h);
        const v8h hi = *(const v8h*)(arow + kb + 16 + 8 * h);
        v16h a;
        #pragma unroll
        for (int t = 0; t < 8; ++t) { a[t] = lo[t]; a[t + 8] = hi[t]; }
        afrag[kc] = a;
    }

    float s[8] = {0.f,0.f,0.f,0.f,0.f,0.f,0.f,0.f};
    const int jBegin = blockIdx.y * COLTILES_PER_CHUNK;
    const _Float16* bBase = z16 + (size_t)r * DIM + 8 * h;

    v16h bufA[KCHUNKS], bufB[KCHUNKS];

    // full-tile B load: 16 consecutive b128 loads (one clause)
    auto loadB = [&](int jt, v16h (&buf)[KCHUNKS]) {
        const _Float16* brow = bBase + (size_t)(jBegin + jt) * 16 * DIM;
        #pragma unroll
        for (int kc = 0; kc < KCHUNKS; ++kc) {
            const v8h blo = *(const v8h*)(brow + kc * 32);
            const v8h bhi = *(const v8h*)(brow + kc * 32 + 16);
            v16h b;
            #pragma unroll
            for (int t = 0; t < 8; ++t) { b[t] = blo[t]; b[t + 8] = bhi[t]; }
            buf[kc] = b;
        }
    };

    // consume one tile: 8 chained WMMAs + scaled/masked exp-sum
    auto compute = [&](int jt, const v16h (&buf)[KCHUNKS]) {
        v8f acc = {};
        #pragma unroll
        for (int kc = 0; kc < KCHUNKS; ++kc) {
            acc = __builtin_amdgcn_wmma_f32_16x16x32_f16(
                      false, afrag[kc], false, buf[kc], (short)0, acc, false, false);
        }
        const int n0 = (jBegin + jt) * 16;
        #pragma unroll
        for (int v = 0; v < 8; ++v) {
            const int row = m0 + v + 8 * h;   // C layout: vgpr v -> row m0+v+8h
            const int col = n0 + r;           //           lane  -> col n0+r
            float x = INV_TEMP * acc[v];
            if (col == row) x = NEG_INF;      // diagonal mask
            s[v] += __expf(x - 2.0f);         // fixed shift: x <= 2 always
        }
    };

    loadB(0, bufA);
    #pragma unroll 1
    for (int jt = 0; jt < COLTILES_PER_CHUNK; jt += 2) {
        loadB(jt + 1, bufB);                  // prefetch next tile
        compute(jt, bufA);                    // consume current
        if (jt + 2 < COLTILES_PER_CHUNK) loadB(jt + 2, bufA);
        compute(jt + 1, bufB);
    }

    // reduce each s[v] across the 16 lanes of this half (same rows)
    #pragma unroll
    for (int v = 0; v < 8; ++v) {
        float t = s[v];
        t += __shfl_xor(t, 1, 32);
        t += __shfl_xor(t, 2, 32);
        t += __shfl_xor(t, 4, 32);
        t += __shfl_xor(t, 8, 32);
        s[v] = t;
    }
    if (r < 8) atomicAdd(&rowsum[m0 + r + 8 * h], s[r]);
}

// ---------------- kernel 4: per-row loss contribution ----------------
// one wave per row i:  contrib = lse_i - p_i
//   lse_i = 2 + log(rowsum_i);  p_i = NEG_INF (i<B) else 2*<z_i, z_{i-B}>
__global__ __launch_bounds__(256)
void ntx_finalize_kernel(const _Float16* __restrict__ z16,
                         const float* __restrict__ rowsum,
                         float* __restrict__ contrib) {
    const int i    = (blockIdx.x * blockDim.x + threadIdx.x) >> 5;
    const int lane = threadIdx.x & 31;
    if (i >= NROWS) return;
    const float lse = 2.0f + __logf(rowsum[i]);
    float p;
    if (i < BATCH) {
        p = NEG_INF;
    } else {
        const v8h va = *(const v8h*)(z16 + (size_t)i * DIM + lane * 8);
        const v8h vb = *(const v8h*)(z16 + (size_t)(i - BATCH) * DIM + lane * 8);
        float d = 0.0f;
        #pragma unroll
        for (int t = 0; t < 8; ++t) d += (float)va[t] * (float)vb[t];
        #pragma unroll
        for (int m = 1; m < 32; m <<= 1) d += __shfl_xor(d, m, 32);
        p = INV_TEMP * d;
    }
    if (lane == 0) contrib[i] = lse - p;
}

// ---------------- kernel 5: deterministic final reduction ----------------
__global__ __launch_bounds__(256)
void ntx_reduce_kernel(const float* __restrict__ contrib,
                       float* __restrict__ out) {
    __shared__ float sm[256];
    float s = 0.0f;
    for (int i = threadIdx.x; i < NROWS; i += 256) s += contrib[i];
    sm[threadIdx.x] = s;
    __syncthreads();
    for (int off = 128; off > 0; off >>= 1) {
        if (threadIdx.x < off) sm[threadIdx.x] += sm[threadIdx.x + off];
        __syncthreads();
    }
    if (threadIdx.x == 0) out[0] = sm[0] / (float)NROWS;
}

// ---------------- launch ----------------
extern "C" void kernel_launch(void* const* d_in, const int* in_sizes, int n_in,
                              void* d_out, int out_size, void* d_ws, size_t ws_size,
                              hipStream_t stream) {
    (void)in_sizes; (void)n_in; (void)out_size; (void)ws_size;
    const float* z1 = (const float*)d_in[0];
    const float* z2 = (const float*)d_in[1];
    float* out = (float*)d_out;

    // workspace layout
    char* w = (char*)d_ws;
    _Float16* z16   = (_Float16*)w;                                  // 8192*256*2 = 4 MB
    float*    rowsum = (float*)(w + (size_t)NROWS * DIM * sizeof(_Float16)); // 32 KB
    float*    contrib = rowsum + NROWS;                               // 32 KB

    ntx_zero_kernel<<<(NROWS + 255) / 256, 256, 0, stream>>>(rowsum);
    ntx_normalize_kernel<<<NROWS / 8, 256, 0, stream>>>(z1, z2, z16);
    ntx_gemm_rowsum_kernel<<<dim3(NROWS / 16, NCHUNK), 32, 0, stream>>>(z16, rowsum);
    ntx_finalize_kernel<<<NROWS / 8, 256, 0, stream>>>(z16, rowsum, contrib);
    ntx_reduce_kernel<<<1, 256, 0, stream>>>(contrib, out);
}
